// GNNModel_47141561041135
// MI455X (gfx1250) — matrix operand order repaired and checked by reading, not compiled, not run on hardware
//
#include <hip/hip_runtime.h>

#define D        128
#define NNODES   10000
#define NEDGES   640000
#define NGRAPHS  16
#define NCLASSES 10

typedef __attribute__((ext_vector_type(16))) __bf16 v16bf;
typedef __attribute__((ext_vector_type(8)))  __bf16 v8bf;
typedef __attribute__((ext_vector_type(8)))  float  v8f;

// ---- workspace layout (bytes, all offsets 256B aligned) ----
static constexpr size_t OFF_XBF   = 0;          // 10000*128*2  = 2,560,000
static constexpr size_t OFF_H     = 2560000;    // 10000*128*4  = 5,120,000
static constexpr size_t OFF_AGG   = 7680000;    // 10000*128*4  = 5,120,000
static constexpr size_t OFF_ZBF   = 12800000;   // 10000*128*2  = 2,560,000
static constexpr size_t OFF_GSUM  = 15360000;   // 16*128*4     = 8,192
static constexpr size_t OFF_GCNT  = 15368192;   // 16*4 (pad 256)
static constexpr size_t OFF_WINPK = 15368448;   // 4*8*32*16*2  = 32,768
static constexpr size_t OFF_WSPK  = 15401216;   // 32,768
static constexpr size_t OFF_WE1PK = 15433984;   // 8*8*32*16*2  = 65,536

__device__ __forceinline__ __bf16 f2bf(float f) {
  unsigned u = __builtin_bit_cast(unsigned, f);
  unsigned r = u + 0x7FFFu + ((u >> 16) & 1u);     // round-to-nearest-even
  unsigned short hs = (unsigned short)(r >> 16);
  return __builtin_bit_cast(__bf16, hs);
}

// A-fragment for V_WMMA_*_16x16x32 (16-bit): lanes 0-15 hold K={0..7,16..23},
// lanes 16-31 hold K={8..15,24..31}; row M = lane%16 (caller supplies row ptr).
__device__ __forceinline__ v16bf load_a_frag(const __bf16* __restrict__ rowptr, int lane) {
  int koff = (lane & 16) ? 8 : 0;
  v8bf lo = *(const v8bf*)(rowptr + koff);
  v8bf hi = *(const v8bf*)(rowptr + koff + 16);
  v16bf a;
#pragma unroll
  for (int i = 0; i < 8; ++i) { a[i] = lo[i]; a[8 + i] = hi[i]; }
  return a;
}

// ---------------- kernel 0: x -> bf16, zero pool accumulators ----------------
__global__ void k_cvt_x(const float* __restrict__ x, __bf16* __restrict__ xbf,
                        float* __restrict__ gsum, float* __restrict__ gcnt) {
  int gid = blockIdx.x * blockDim.x + threadIdx.x;
  if (gid < NNODES * D) xbf[gid] = f2bf(x[gid]);
  if (gid < NGRAPHS * D) gsum[gid] = 0.0f;
  if (gid < NGRAPHS)     gcnt[gid] = 0.0f;
}

// ---------------- kernel 1: pack weights to WMMA B layout ----------------
// packed[chunk(kc*8+nt)][lane(32)][half(16)], B layout: lane holds column
// n = nt*16 + lane%16; half h <-> K = kc*32 + (lane<16 ? 0 : 8) + (h&7) + (h<8 ? 0 : 16).
__global__ void k_pack_w(const float* __restrict__ Win, const float* __restrict__ Wself,
                         const float* __restrict__ We1, __bf16* __restrict__ winpk,
                         __bf16* __restrict__ wspk, __bf16* __restrict__ we1pk) {
  int gid = blockIdx.x * blockDim.x + threadIdx.x;   // 65536 threads
  const float* src; __bf16* dst; int p;
  if (gid < 16384)      { src = Win;   dst = winpk; p = gid;         }
  else if (gid < 32768) { src = Wself; dst = wspk;  p = gid - 16384; }
  else                  { src = We1;   dst = we1pk; p = gid - 32768; }
  int h     = p & 15;
  int lane  = (p >> 4) & 31;
  int chunk = p >> 9;              // kc*8 + nt
  int nt = chunk & 7, kc = chunk >> 3;
  int n = nt * 16 + (lane & 15);
  int k = kc * 32 + ((lane & 16) ? 8 : 0) + (h & 7) + ((h & 8) ? 16 : 0);
  dst[p] = f2bf(src[k * D + n]);   // W is row-major [K][D]
}

// ---------------- kernel 2: node GEMMs: h = x@W_in ; agg = x@W_self ----------------
__global__ void __launch_bounds__(32) k_node_gemm(
    const __bf16* __restrict__ xbf, const __bf16* __restrict__ winpk,
    const __bf16* __restrict__ wspk, float* __restrict__ h, float* __restrict__ agg) {
  int tile = blockIdx.x;                 // 625 tiles * 16 rows = 10000 exact
  int lane = threadIdx.x;
  int m = tile * 16 + (lane & 15);
  const __bf16* row = xbf + (size_t)m * D;

  v16bf A[4];
#pragma unroll
  for (int kc = 0; kc < 4; ++kc) A[kc] = load_a_frag(row + kc * 32, lane);

  int crow = (lane & 16) ? 8 : 0;
  int ccol = lane & 15;
#pragma unroll
  for (int wt = 0; wt < 2; ++wt) {
    const __bf16* bpk = wt ? wspk : winpk;
    float* outbuf = wt ? agg : h;
#pragma unroll
    for (int nt = 0; nt < 8; ++nt) {
      v8f acc = {};
#pragma unroll
      for (int kc = 0; kc < 4; ++kc) {
        v16bf b = *(const v16bf*)(bpk + ((size_t)((kc * 8 + nt) * 32 + lane)) * 16);
        acc = __builtin_amdgcn_wmma_f32_16x16x32_bf16(false, A[kc], false, b,
                                                      (short)0, acc, false, false);
      }
      int n = nt * 16 + ccol;
#pragma unroll
      for (int r = 0; r < 8; ++r)
        outbuf[(size_t)(tile * 16 + crow + r) * D + n] = acc[r];
    }
  }
}

// ---------------- kernel 3: scatter-add agg[dst] += h[src] ----------------
__global__ void k_scatter(const float* __restrict__ h, const int* __restrict__ ei,
                          float* __restrict__ agg) {
  long gid = (long)blockIdx.x * blockDim.x + threadIdx.x;
  int e = (int)(gid >> 7);
  int d = (int)(gid & 127);
  int s = ei[e];
  int t = ei[NEDGES + e];
  atomicAdd(&agg[(size_t)t * D + d], h[(size_t)s * D + d]);
}

// ---------------- kernel 4: z = relu(agg + b); pool sums; z -> bf16 ----------------
__global__ void k_znode(const float* __restrict__ agg, const float* __restrict__ b,
                        const int* __restrict__ batch, __bf16* __restrict__ zbf,
                        float* __restrict__ gsum, float* __restrict__ gcnt) {
  int gid = blockIdx.x * blockDim.x + threadIdx.x;
  int node = gid >> 7, d = gid & 127;
  float z = agg[gid] + b[d];
  z = z > 0.f ? z : 0.f;
  zbf[gid] = f2bf(z);
  int g = batch[node];
  atomicAdd(&gsum[g * D + d], z);
  if (d == 0) atomicAdd(&gcnt[g], 1.0f);
}

// ---------------- kernel 5: edge MLP (dominant GEMM, WMMA + async LDS) ----------------
// per wave: one 16-edge tile, K=256 (src||dst features), N=128, then relu+dot(W_e2)
__global__ void __launch_bounds__(256) k_edge_mlp(
    const __bf16* __restrict__ zbf, const __bf16* __restrict__ we1pk,
    const int* __restrict__ ei, const float* __restrict__ b1,
    const float* __restrict__ w2, const float* __restrict__ b2,
    float* __restrict__ out) {
  __shared__ v16bf sW[2048];                         // packed W_e1, 64 KB
  int t = threadIdx.x;

  // Async global->LDS fill of packed W_e1: 256 B/thread = 16 x b128 transfers,
  // written directly to LDS by the async unit (no VGPR staging), tracked by
  // ASYNCcnt. Flat LDS addresses are {SHARED_BASE, offset[31:0]}, so the low
  // 32 bits of the shared pointer are the wave-relative LDS byte address.
  {
    unsigned long long ga0 =
        (unsigned long long)(const char*)we1pk + (unsigned)(t * 16);
    unsigned lds0 =
        (unsigned)(unsigned long long)(const char*)(&sW[0]) + (unsigned)(t * 16);
#pragma unroll
    for (int i = 0; i < 16; ++i) {
      unsigned      ldso = lds0 + (unsigned)(i * 256 * 16);
      unsigned long long ga = ga0 + (unsigned long long)(i * 256 * 16);
      asm volatile("global_load_async_to_lds_b128 %0, %1, off"
                   :: "v"(ldso), "v"(ga) : "memory");
    }
    asm volatile("s_wait_asynccnt 0x0" ::: "memory");
  }
  __syncthreads();

  int lane = t & 31;
  int wave = t >> 5;
  int tile = blockIdx.x * 8 + wave;                  // 5000*8 = 40000 tiles exact
  int m = lane & 15;
  int e = tile * 16 + m;
  int s  = ei[e];
  int dd = ei[NEDGES + e];
  const __bf16* srow = zbf + (size_t)s  * D;
  const __bf16* drow = zbf + (size_t)dd * D;

  v16bf A[8];                                        // K=256 -> 8 chunks of 32
#pragma unroll
  for (int kc = 0; kc < 8; ++kc) {
    const __bf16* row = (kc < 4) ? srow : drow;
    A[kc] = load_a_frag(row + (kc & 3) * 32, lane);
  }

  float partial[8];
#pragma unroll
  for (int r = 0; r < 8; ++r) partial[r] = 0.f;

#pragma unroll
  for (int nt = 0; nt < 8; ++nt) {
    v8f acc = {};
#pragma unroll
    for (int kc = 0; kc < 8; ++kc) {
      v16bf b = sW[(kc * 8 + nt) * 32 + lane];
      acc = __builtin_amdgcn_wmma_f32_16x16x32_bf16(false, A[kc], false, b,
                                                    (short)0, acc, false, false);
    }
    int col = nt * 16 + m;
    float bias = b1[col];
    float w = w2[col];
#pragma unroll
    for (int r = 0; r < 8; ++r) {
      float hv = acc[r] + bias;
      hv = hv > 0.f ? hv : 0.f;
      partial[r] += hv * w;                          // fold @W_e2 into reduction
    }
  }
  // reduce the 16 columns held across each 16-lane group
#pragma unroll
  for (int r = 0; r < 8; ++r) {
    float v = partial[r];
    v += __shfl_xor(v, 1, 16);
    v += __shfl_xor(v, 2, 16);
    v += __shfl_xor(v, 4, 16);
    v += __shfl_xor(v, 8, 16);
    partial[r] = v;
  }
  float bias2 = b2[0];
  if ((lane & 15) == 0) {
    int rowbase = tile * 16 + ((lane & 16) ? 8 : 0); // lanes 0-15: M=0..7, 16-31: M=8..15
#pragma unroll
    for (int r = 0; r < 8; ++r) out[rowbase + r] = partial[r] + bias2;
  }
}

// ---------------- kernel 6: graph classifier ----------------
__global__ void k_cls(const float* __restrict__ gsum, const float* __restrict__ gcnt,
                      const float* __restrict__ Wc, const float* __restrict__ bc,
                      float* __restrict__ out) {
  int t = blockIdx.x * blockDim.x + threadIdx.x;
  if (t >= NGRAPHS * NCLASSES) return;
  int g = t / NCLASSES, c = t % NCLASSES;
  float cnt = gcnt[g]; cnt = cnt > 1.f ? cnt : 1.f;
  float inv = 1.f / cnt;
  float acc = bc[c];
  for (int d0 = 0; d0 < D; ++d0)
    acc += gsum[g * D + d0] * inv * Wc[d0 * NCLASSES + c];
  out[NEDGES + t] = acc;
}

extern "C" void kernel_launch(void* const* d_in, const int* in_sizes, int n_in,
                              void* d_out, int out_size, void* d_ws, size_t ws_size,
                              hipStream_t stream) {
  (void)in_sizes; (void)n_in; (void)out_size; (void)ws_size;
  const float* x      = (const float*)d_in[0];
  const int*   ei     = (const int*)d_in[1];
  const int*   batch  = (const int*)d_in[2];
  const float* W_in   = (const float*)d_in[3];
  const float* W_self = (const float*)d_in[4];
  const float* b      = (const float*)d_in[5];
  const float* W_e1   = (const float*)d_in[6];
  const float* b_e1   = (const float*)d_in[7];
  const float* W_e2   = (const float*)d_in[8];
  const float* b_e2   = (const float*)d_in[9];
  const float* W_cls  = (const float*)d_in[10];
  const float* b_cls  = (const float*)d_in[11];
  float* out = (float*)d_out;
  char*  ws  = (char*)d_ws;

  __bf16* xbf   = (__bf16*)(ws + OFF_XBF);
  float*  h     = (float*) (ws + OFF_H);
  float*  agg   = (float*) (ws + OFF_AGG);
  __bf16* zbf   = (__bf16*)(ws + OFF_ZBF);
  float*  gsum  = (float*) (ws + OFF_GSUM);
  float*  gcnt  = (float*) (ws + OFF_GCNT);
  __bf16* winpk = (__bf16*)(ws + OFF_WINPK);
  __bf16* wspk  = (__bf16*)(ws + OFF_WSPK);
  __bf16* we1pk = (__bf16*)(ws + OFF_WE1PK);

  k_cvt_x   <<<(NNODES * D + 255) / 256, 256, 0, stream>>>(x, xbf, gsum, gcnt);
  k_pack_w  <<<65536 / 256, 256, 0, stream>>>(W_in, W_self, W_e1, winpk, wspk, we1pk);
  k_node_gemm<<<NNODES / 16, 32, 0, stream>>>(xbf, winpk, wspk, h, agg);
  k_scatter <<<(NEDGES * 128) / 256, 256, 0, stream>>>(h, ei, agg);
  k_znode   <<<(NNODES * D) / 256, 256, 0, stream>>>(agg, b, batch, zbf, gsum, gcnt);
  k_edge_mlp<<<NEDGES / 16 / 8, 256, 0, stream>>>(zbf, we1pk, ei, b_e1, W_e2, b_e2, out);
  k_cls     <<<1, 256, 0, stream>>>(gsum, gcnt, W_cls, b_cls, out);
}
